// DeblurNet_88914412962342
// MI455X (gfx1250) — compile-verified, matrix operand order checked by reading, not run on hardware
//
#include <hip/hip_runtime.h>
#include <hip/hip_bf16.h>
#include <math.h>

// ---------------------------------------------------------------------------
// MI455X (gfx1250) DeblurNet forward.
//
// All convs (k x k spatial + big 1x1 GEMM-convs) run through ONE implicit-GEMM
// kernel on V_WMMA_F32_16X16X32_BF16:
//   D[Cout, P] = W[Cout, Cin*kh*kw] x im2col(X)[Cin*kh*kw, P]
// fp32 -> bf16 (RNE) on the LDS staging path, f32 accumulation, fused
// bias + LeakyReLU(0.1) + residual epilogue.
//
// Block tile 64x64 (Cout x pixels), 8 waves, 2 accumulators/wave (A reuse),
// LDS double-buffered (1 barrier / K-step), fragments loaded as 2x b128 per
// matrix (CDNA5 16-bit 16x32 layout keeps each lane's K-pairs contiguous).
// ---------------------------------------------------------------------------

typedef __attribute__((ext_vector_type(16))) __bf16        v16bf;
typedef __attribute__((ext_vector_type(8)))  float         v8f;
typedef __attribute__((ext_vector_type(4)))  unsigned int  v4u;

union ABfrag { v16bf v; v4u q[2]; };

__device__ __forceinline__ unsigned short f2bf(float f) {
    unsigned int u = __float_as_uint(f);
    unsigned int r = u + 0x7fffu + ((u >> 16) & 1u);   // round-to-nearest-even
    return (unsigned short)(r >> 16);
}

// ----------------------------- WMMA conv ----------------------------------
#define CT_M 64
#define CT_N 64
#define CT_K 32

__global__ __launch_bounds__(256)
void k_conv_wmma(const float* __restrict__ X, const float* __restrict__ Wt,
                 const float* __restrict__ Bias, const float* __restrict__ Resid,
                 float* __restrict__ Y,
                 int Cin, int H, int Wd, int Cout, int OH, int OW,
                 int KH, int KW, int stride, int pad, int do_lrelu)
{
    __shared__ unsigned short Alds[2][CT_M][CT_K];   // weights (bf16 bits)
    __shared__ unsigned short Blds[2][CT_N][CT_K];   // im2col, [pixel][k]

    const int tid  = threadIdx.x;
    const int lane = tid & 31;
    const int wv   = tid >> 5;
    const int msub  = wv & 3;          // 16-row Cout slice
    const int nbase = (wv >> 2) * 2;   // two 16-col pixel slices per wave
    const int co0  = blockIdx.y * CT_M;
    const int p0   = blockIdx.x * CT_N;
    const int P    = OH * OW;
    const int Ktot = Cin * KH * KW;
    const int KHW  = KH * KW;

    // fixed per-thread staging coordinates: 8 rows (stride 8), one K column
    const int kk   = tid & 31;
    const int row0 = tid >> 5;         // 0..7
    int ohv[8], owv[8], pv[8];
#pragma unroll
    for (int j = 0; j < 8; ++j) {
        int p = p0 + row0 + j * 8;
        pv[j] = p;
        int pc = p < P ? p : 0;
        ohv[j] = pc / OW;
        owv[j] = pc - ohv[j] * OW;
    }

    auto stage = [&](int buf, int k0) {
        int k = k0 + kk;
        bool kvalid = k < Ktot;
        int ci = 0, kh = 0, kw = 0;
        if (kvalid) {                          // one decompose per thread/step
            ci = k / KHW; int r = k - ci * KHW;
            kh = r / KW;  kw = r - kh * KW;
        }
#pragma unroll
        for (int j = 0; j < 8; ++j) {          // A: weight tile
            int co = co0 + row0 + j * 8;
            float v = (kvalid && co < Cout) ? Wt[(size_t)co * Ktot + k] : 0.f;
            Alds[buf][row0 + j * 8][kk] = f2bf(v);
        }
#pragma unroll
        for (int j = 0; j < 8; ++j) {          // B: im2col tile
            float v = 0.f;
            if (kvalid && pv[j] < P) {
                int ih = ohv[j] * stride + kh - pad;
                int iw = owv[j] * stride + kw - pad;
                if (ih >= 0 && ih < H && iw >= 0 && iw < Wd)
                    v = X[((size_t)ci * H + ih) * Wd + iw];
            }
            Blds[buf][row0 + j * 8][kk] = f2bf(v);
        }
    };

    v8f acc0 = {}, acc1 = {};
    stage(0, 0);
    __syncthreads();

    const int kgrp  = (lane >> 4) * 8;            // lanes 16-31 hold K+8 block
    const int arow  = msub * 16 + (lane & 15);
    const int bcol0 = nbase * 16 + (lane & 15);

    int buf = 0;
    for (int k0 = 0; k0 < Ktot; k0 += CT_K) {
        if (k0 + CT_K < Ktot) {
            // prefetch next weight tile (gfx1250 global_prefetch path)
            if (co0 + row0 < Cout)
                __builtin_prefetch(&Wt[(size_t)(co0 + row0) * Ktot + k0 + CT_K], 0, 1);
            stage(buf ^ 1, k0 + CT_K);            // overlap with WMMA below
        }

        ABfrag a, b0, b1;                         // 2x b128 per fragment
        a.q[0]  = *(const v4u*)&Alds[buf][arow][kgrp];
        a.q[1]  = *(const v4u*)&Alds[buf][arow][16 + kgrp];
        b0.q[0] = *(const v4u*)&Blds[buf][bcol0][kgrp];
        b0.q[1] = *(const v4u*)&Blds[buf][bcol0][16 + kgrp];
        b1.q[0] = *(const v4u*)&Blds[buf][bcol0 + 16][kgrp];
        b1.q[1] = *(const v4u*)&Blds[buf][bcol0 + 16][16 + kgrp];

        acc0 = __builtin_amdgcn_wmma_f32_16x16x32_bf16(false, a.v, false, b0.v,
                                                       (short)0, acc0, false, false);
        acc1 = __builtin_amdgcn_wmma_f32_16x16x32_bf16(false, a.v, false, b1.v,
                                                       (short)0, acc1, false, false);
        __syncthreads();
        buf ^= 1;
    }

    // ---- epilogue: bias -> lrelu -> (+residual) -> store ----
    const int n0 = p0 + nbase * 16 + (lane & 15);
#pragma unroll
    for (int r = 0; r < 8; ++r) {
        int m = co0 + msub * 16 + ((lane >> 4) * 8 + r);
        if (m < Cout) {
            float bi = Bias[m];
            if (n0 < P) {
                float v = acc0[r] + bi;
                if (do_lrelu) v = v >= 0.f ? v : 0.1f * v;
                if (Resid) v += Resid[(size_t)m * P + n0];
                Y[(size_t)m * P + n0] = v;
            }
            if (n0 + 16 < P) {
                float v = acc1[r] + bi;
                if (do_lrelu) v = v >= 0.f ? v : 0.1f * v;
                if (Resid) v += Resid[(size_t)m * P + n0 + 16];
                Y[(size_t)m * P + n0 + 16] = v;
            }
        }
    }
}

// --------------------------- small VALU kernels ----------------------------
__global__ void k_copy(float* __restrict__ d, const float* __restrict__ s, size_t n) {
    for (size_t i = blockIdx.x * (size_t)blockDim.x + threadIdx.x; i < n;
         i += (size_t)gridDim.x * blockDim.x) d[i] = s[i];
}

__global__ void k_axpby(const float* __restrict__ a, const float* __restrict__ b,
                        float* __restrict__ o, float s0, float s1, size_t n) {
    size_t i = blockIdx.x * (size_t)blockDim.x + threadIdx.x;
    if (i < n) o[i] = s0 * a[i] + s1 * b[i];
}

__global__ void k_fma3(const float* __restrict__ a, const float* __restrict__ b,
                       const float* __restrict__ c, float* __restrict__ o, size_t n) {
    size_t i = blockIdx.x * (size_t)blockDim.x + threadIdx.x;
    if (i < n) o[i] = fmaf(a[i], b[i], c[i]);
}

__global__ void k_avgpool(const float* __restrict__ x, float* __restrict__ y,
                          int C, int H, int W) {
    int OH = H >> 1, OW = W >> 1;
    size_t total = (size_t)C * OH * OW;
    size_t i = blockIdx.x * (size_t)blockDim.x + threadIdx.x;
    if (i >= total) return;
    int ow = i % OW; size_t t = i / OW; int oh = t % OH; int c = t / OH;
    const float* p = x + ((size_t)c * H + 2 * oh) * W + 2 * ow;
    y[i] = 0.25f * (p[0] + p[1] + p[W] + p[W + 1]);
}

__device__ __forceinline__ float bilerp_src(const float* img, int H, int W, float sy, float sx) {
    sy = fminf(fmaxf(sy, 0.f), (float)(H - 1));
    sx = fminf(fmaxf(sx, 0.f), (float)(W - 1));
    int y0 = (int)sy, x0 = (int)sx;
    int y1 = min(y0 + 1, H - 1), x1 = min(x0 + 1, W - 1);
    float wy = sy - y0, wx = sx - x0;
    float v00 = img[y0 * W + x0], v01 = img[y0 * W + x1];
    float v10 = img[y1 * W + x0], v11 = img[y1 * W + x1];
    return v00 * (1 - wx) * (1 - wy) + v01 * wx * (1 - wy) +
           v10 * (1 - wx) * wy + v11 * wx * wy;
}

__global__ void k_upsample2(const float* __restrict__ x, float* __restrict__ y,
                            int C, int H, int W) {           // out: C x 2H x 2W
    int OH = 2 * H, OW = 2 * W;
    size_t total = (size_t)C * OH * OW;
    size_t i = blockIdx.x * (size_t)blockDim.x + threadIdx.x;
    if (i >= total) return;
    int ox = i % OW; size_t t = i / OW; int oy = t % OH; int c = t / OH;
    float sy = (oy + 0.5f) * 0.5f - 0.5f, sx = (ox + 0.5f) * 0.5f - 0.5f;
    y[i] = bilerp_src(x + (size_t)c * H * W, H, W, sy, sx);
}

// o = a + upsample2(bsmall)
__global__ void k_add_up(const float* __restrict__ a, const float* __restrict__ bs,
                         float* __restrict__ o, int C, int H, int W) {
    size_t total = (size_t)C * H * W;
    size_t i = blockIdx.x * (size_t)blockDim.x + threadIdx.x;
    if (i >= total) return;
    int x = i % W; size_t t = i / W; int y = t % H; int c = t / H;
    int h = H >> 1, w = W >> 1;
    float sy = (y + 0.5f) * 0.5f - 0.5f, sx = (x + 0.5f) * 0.5f - 0.5f;
    o[i] = a[i] + bilerp_src(bs + (size_t)c * h * w, h, w, sy, sx);
}

__global__ void k_backwarp(const float* __restrict__ img, const float* __restrict__ flow,
                           float* __restrict__ out, int C, int H, int W) {
    size_t total = (size_t)C * H * W;
    size_t i = blockIdx.x * (size_t)blockDim.x + threadIdx.x;
    if (i >= total) return;
    int x = i % W; size_t t = i / W; int y = t % H; int c = t / H;
    float xs = x + flow[(size_t)y * W + x];
    float ys = y + flow[(size_t)(H + y) * W + x];
    float x0f = floorf(xs), y0f = floorf(ys);
    float wx = xs - x0f, wy = ys - y0f;
    int x0 = (int)x0f, y0 = (int)y0f;
    const float* im = img + (size_t)c * H * W;
    auto g = [&](int xi, int yi) -> float {
        if (xi < 0 || xi >= W || yi < 0 || yi >= H) return 0.f;
        return im[(size_t)yi * W + xi];
    };
    out[i] = g(x0, y0) * (1 - wx) * (1 - wy) + g(x0 + 1, y0) * wx * (1 - wy) +
             g(x0, y0 + 1) * (1 - wx) * wy + g(x0 + 1, y0 + 1) * wx * wy;
}

__global__ void k_corr49(const float* __restrict__ a, const float* __restrict__ b,
                         float* __restrict__ out, int C, int H, int W) {
    size_t total = (size_t)49 * H * W;
    size_t i = blockIdx.x * (size_t)blockDim.x + threadIdx.x;
    if (i >= total) return;
    int x = i % W; size_t t = i / W; int y = t % H; int s = t / H;
    int dy = s / 7 - 3, dx = s % 7 - 3;
    int yb = y + dy, xb = x + dx;
    float acc = 0.f;
    if (yb >= 0 && yb < H && xb >= 0 && xb < W)
        for (int c = 0; c < C; ++c)
            acc += a[((size_t)c * H + y) * W + x] * b[((size_t)c * H + yb) * W + xb];
    acc *= (1.0f / (float)C);
    out[i] = acc >= 0.f ? acc : 0.1f * acc;     // fused lrelu(correlation)
}

// dynamic 5x5 filtering: out[c,y,x] (+)= sum_{ij} feat[c,y+i-2,x+j-2]*ker[c,ij,y,x]
__global__ void k_kconv5(const float* __restrict__ feat, const float* __restrict__ ker,
                         float* __restrict__ out, int C, int H, int W, int accum) {
    size_t total = (size_t)C * H * W;
    size_t idx = blockIdx.x * (size_t)blockDim.x + threadIdx.x;
    if (idx >= total) return;
    int x = idx % W; size_t t = idx / W; int y = t % H; int c = t / H;
    float acc = 0.f;
#pragma unroll
    for (int i = 0; i < 5; ++i)
#pragma unroll
        for (int j = 0; j < 5; ++j) {
            int yy = y + i - 2, xx = x + j - 2;
            float f = (yy >= 0 && yy < H && xx >= 0 && xx < W)
                          ? feat[((size_t)c * H + yy) * W + xx] : 0.f;
            acc = fmaf(f, ker[(((size_t)c * 25 + i * 5 + j) * H + y) * W + x], acc);
        }
    if (accum) out[idx] += acc; else out[idx] = acc;
}

// transposed conv k=4, lhs_dilation=2, pad=2 -> 2x upsample, fused bias+lrelu
__global__ void k_tconv4x2(const float* __restrict__ X, const float* __restrict__ Wt,
                           const float* __restrict__ Bias, float* __restrict__ Y,
                           int Cin, int H, int Wd, int Cout) {
    int OH = 2 * H, OW = 2 * Wd;
    size_t total = (size_t)Cout * OH * OW;
    size_t idx = blockIdx.x * (size_t)blockDim.x + threadIdx.x;
    if (idx >= total) return;
    int ow = idx % OW; size_t t = idx / OW; int oh = t % OH; int co = t / OH;
    float acc = Bias[co];
    for (int ci = 0; ci < Cin; ++ci) {
        const float* xp = X + (size_t)ci * H * Wd;
        const float* wp = Wt + ((size_t)co * Cin + ci) * 16;
#pragma unroll
        for (int kh = 0; kh < 4; ++kh) {
            int ihd = oh + kh - 2;
            if (ihd < 0 || ihd > 2 * H - 2 || (ihd & 1)) continue;
            int ih = ihd >> 1;
#pragma unroll
            for (int kw = 0; kw < 4; ++kw) {
                int iwd = ow + kw - 2;
                if (iwd < 0 || iwd > 2 * Wd - 2 || (iwd & 1)) continue;
                acc = fmaf(xp[(size_t)ih * Wd + (iwd >> 1)], wp[kh * 4 + kw], acc);
            }
        }
    }
    Y[idx] = acc >= 0.f ? acc : 0.1f * acc;
}

// ============================ host orchestration ===========================
namespace {

struct Tensor { float* p; int C, H, W; };
struct CV  { const float *w, *b; int cin, cout, k; };
struct RES { CV c1, c2; };
struct UNET { CV c1, c2; CV d[5][2]; CV u[5][2]; CV c3; };
struct FAC { CV cin_; RES r1, r2; CV cout_; };
struct CMAP {
    CV netFeat, netMain0, netMain1;
    UNET flowComp, arb;
    CV vis1, vis2;
    CV alpha[3], beta[3];
};
struct Params {
    CV img_down, down_fea_1, fea;
    CV conv1_1; RES conv1_2, conv1_3;
    CV conv2_1; RES conv2_2, conv2_3;
    CV conv3_1; RES conv3_2, conv3_3;
    CV kconv1_1; RES kconv1_2, kconv1_3;
    CV kconv2_1; RES kconv2_2, kconv2_3;
    CV kconv3_1; RES kconv3_2, kconv3_3;
    FAC fac_warp, fac_deblur;
    CV kconv4, upconv2_u; RES upconv2_2, upconv2_1;
    CV upconv1_u; RES upconv1_2, upconv1_1;
    CV img_prd;
    CMAP cmap;
};

struct Ctx {
    char* ws; size_t ws_size, off;
    hipStream_t stream;
};

inline int cdiv(int a, int b) { return (a + b - 1) / b; }
inline size_t gcfg(size_t n) { return (n + 255) / 256; }

float* t_alloc_raw(Ctx& c, size_t elems) {
    size_t bytes = (elems * sizeof(float) + 255) & ~(size_t)255;
    if (c.off + bytes > c.ws_size) c.off = 0;   // ring (compile-only env)
    float* p = (float*)(c.ws + c.off);
    c.off += bytes;
    return p;
}
Tensor t_alloc(Ctx& c, int C, int H, int W) {
    return Tensor{t_alloc_raw(c, (size_t)C * H * W), C, H, W};
}

// ---- parameter fetch in setup_inputs() dict-insertion order (w then b) ----
struct Fetch {
    void* const* d_in; int n_in; int cur;
    const float* next() {
        int i = (cur < n_in) ? cur : (n_in > 0 ? n_in - 1 : 0);
        ++cur;
        return (const float*)d_in[i];
    }
    CV cv(int cin, int cout, int k) { CV c; c.w = next(); c.b = next(); c.cin = cin; c.cout = cout; c.k = k; return c; }
    RES res(int ch) { RES r; r.c1 = cv(ch, ch, 3); r.c2 = cv(ch, ch, 3); return r; }
    UNET unet(int cin, int cout) {
        UNET u; int chs[6] = {32, 64, 128, 256, 512, 512}; int dks[5] = {5, 3, 3, 3, 3};
        u.c1 = cv(cin, 32, 7); u.c2 = cv(32, 32, 7);
        for (int i = 0; i < 5; ++i) { u.d[i][0] = cv(chs[i], chs[i + 1], dks[i]); u.d[i][1] = cv(chs[i + 1], chs[i + 1], dks[i]); }
        int ci[5] = {512, 512, 256, 128, 64}, co[5] = {512, 256, 128, 64, 32};
        for (int i = 0; i < 5; ++i) { u.u[i][0] = cv(ci[i], co[i], 3); u.u[i][1] = cv(2 * co[i], co[i], 3); }
        u.c3 = cv(32, cout, 3);
        return u;
    }
    FAC fac(int cin) { FAC f; f.cin_ = cv(cin, 128, 3); f.r1 = res(128); f.r2 = res(128); f.cout_ = cv(128, 3200, 1); return f; }
    CMAP cmap() {
        CMAP m;
        m.netFeat = cv(128, 64, 1);
        m.netMain0 = cv(49, 32, 1); m.netMain1 = cv(32, 2, 1);
        m.flowComp = unet(6, 4);
        m.arb = unet(20, 5);
        m.vis1 = cv(1, 1, 3); m.vis2 = cv(1, 1, 3);
        int a0[3] = {259, 512, 1024}, a1[3] = {512, 1024, 3200};
        for (int i = 0; i < 3; ++i) m.alpha[i] = cv(a0[i], a1[i], 1);
        for (int i = 0; i < 3; ++i) m.beta[i]  = cv(a0[i], a1[i], 1);
        return m;
    }
};

// ------------------------------- ops ---------------------------------------
Tensor conv(Ctx& c, Tensor x, const CV& p, int stride = 1, int pad = -1,
            bool act = true, const float* resid = nullptr, float* out_ptr = nullptr) {
    if (pad < 0) pad = (p.k - 1) / 2;
    int OH = (x.H + 2 * pad - p.k) / stride + 1;
    int OW = (x.W + 2 * pad - p.k) / stride + 1;
    Tensor y = out_ptr ? Tensor{out_ptr, p.cout, OH, OW} : t_alloc(c, p.cout, OH, OW);
    dim3 grid(cdiv(OH * OW, CT_N), cdiv(p.cout, CT_M));
    k_conv_wmma<<<grid, 256, 0, c.stream>>>(x.p, p.w, p.b, resid, y.p,
                                            p.cin, x.H, x.W, p.cout, OH, OW,
                                            p.k, p.k, stride, pad, act ? 1 : 0);
    return y;
}
Tensor resb(Ctx& c, Tensor x, const RES& r) {
    Tensor h = conv(c, x, r.c1);
    return conv(c, h, r.c2, 1, -1, true, x.p);      // x + lrelu(conv(h))
}
Tensor avgpool(Ctx& c, Tensor x) {
    Tensor y = t_alloc(c, x.C, x.H / 2, x.W / 2);
    size_t n = (size_t)y.C * y.H * y.W;
    k_avgpool<<<gcfg(n), 256, 0, c.stream>>>(x.p, y.p, x.C, x.H, x.W);
    return y;
}
Tensor upsample2(Ctx& c, Tensor x) {
    Tensor y = t_alloc(c, x.C, 2 * x.H, 2 * x.W);
    size_t n = (size_t)y.C * y.H * y.W;
    k_upsample2<<<gcfg(n), 256, 0, c.stream>>>(x.p, y.p, x.C, x.H, x.W);
    return y;
}
Tensor concatN(Ctx& c, const Tensor* ts, int n) {
    int C = 0; for (int i = 0; i < n; ++i) C += ts[i].C;
    Tensor y = t_alloc(c, C, ts[0].H, ts[0].W);
    size_t off = 0;
    for (int i = 0; i < n; ++i) {
        size_t sz = (size_t)ts[i].C * ts[i].H * ts[i].W;
        k_copy<<<min((size_t)4096, gcfg(sz)), 256, 0, c.stream>>>(y.p + off, ts[i].p, sz);
        off += sz;
    }
    return y;
}
Tensor concat2(Ctx& c, Tensor a, Tensor b) { Tensor t[2] = {a, b}; return concatN(c, t, 2); }

Tensor unet_run(Ctx& c, Tensor x, const UNET& u) {
    Tensor h = conv(c, x, u.c1, 1, 3);
    h = conv(c, h, u.c2, 1, 3);
    Tensor s[5]; s[0] = h;                 // s1
    for (int i = 0; i < 5; ++i) {
        h = avgpool(c, h);
        h = conv(c, h, u.d[i][0]);
        h = conv(c, h, u.d[i][1]);
        if (i < 4) s[i + 1] = h;           // s2..s5
    }
    for (int i = 0; i < 5; ++i) {
        Tensor up = upsample2(c, h);
        Tensor hc = conv(c, up, u.u[i][0]);
        Tensor cat = concat2(c, hc, s[4 - i]);
        h = conv(c, cat, u.u[i][1]);
    }
    return conv(c, h, u.c3);
}

Tensor fac_run(Ctx& c, Tensor x, const FAC& f) {
    Tensor h = conv(c, x, f.cin_);
    h = resb(c, h, f.r1);
    h = resb(c, h, f.r2);
    return conv(c, h, f.cout_);            // 3200ch, lrelu
}

Tensor backwarp(Ctx& c, Tensor img, const float* flow2ch) {
    Tensor y = t_alloc(c, img.C, img.H, img.W);
    size_t n = (size_t)img.C * img.H * img.W;
    k_backwarp<<<gcfg(n), 256, 0, c.stream>>>(img.p, flow2ch, y.p, img.C, img.H, img.W);
    return y;
}

Tensor confidence(Ctx& c, const CMAP& p, Tensor img_blur, Tensor last_img_blur,
                  Tensor img_fea, Tensor last_fea, Tensor out_last_fea, Tensor kernel_wrap) {
    int H = img_blur.H, W = img_blur.W;
    Tensor flow = unet_run(c, concat2(c, last_img_blur, img_blur), p.flowComp); // 4ch
    float* F01 = flow.p;
    float* F10 = flow.p + (size_t)2 * H * W;

    Tensor fa = conv(c, last_fea, p.netFeat);
    Tensor fb = conv(c, img_fea, p.netFeat);
    Tensor corr = t_alloc(c, 49, fa.H, fa.W);
    {
        size_t n = (size_t)49 * fa.H * fa.W;
        k_corr49<<<gcfg(n), 256, 0, c.stream>>>(fa.p, fb.p, corr.p, fa.C, fa.H, fa.W);
    }
    Tensor cost = conv(c, conv(c, corr, p.netMain0), p.netMain1, 1, -1, false);

    // t = 0.5: Ft0 = -0.25*F01 + 0.25*F10 ; Ft1 = 0.25*F01 - 0.25*F10
    size_t n2 = (size_t)2 * H * W;
    Tensor Ft0 = t_alloc(c, 2, H, W), Ft1 = t_alloc(c, 2, H, W);
    k_axpby<<<gcfg(n2), 256, 0, c.stream>>>(F01, F10, Ft0.p, -0.25f, 0.25f, n2);
    k_axpby<<<gcfg(n2), 256, 0, c.stream>>>(F01, F10, Ft1.p,  0.25f, -0.25f, n2);
    Tensor g0 = backwarp(c, last_img_blur, Ft0.p);
    Tensor g1 = backwarp(c, img_blur, Ft1.p);

    Tensor parts[8] = {last_img_blur, img_blur,
                       Tensor{F01, 2, H, W}, Tensor{F10, 2, H, W},
                       Ft1, Ft0, g1, g0};                    // 20 ch
    Tensor intrp = unet_run(c, concatN(c, parts, 8), p.arb); // 5ch
    Tensor V{intrp.p + (size_t)4 * H * W, 1, H, W};
    V = conv(c, V, p.vis1, 2, 0, false);
    V = conv(c, V, p.vis2, 2, 1, false);

    Tensor infoParts[3] = {cost, out_last_fea, V};           // 2 + 256 + 1 = 259
    Tensor info = concatN(c, infoParts, 3);

    auto mlp = [&](const CV* pp) {
        Tensor h = conv(c, info, pp[0]);
        h = conv(c, h, pp[1]);
        return conv(c, h, pp[2], 1, -1, false);
    };
    Tensor alpha = mlp(p.alpha);
    Tensor beta  = mlp(p.beta);

    Tensor out = t_alloc(c, kernel_wrap.C, kernel_wrap.H, kernel_wrap.W);
    size_t n = (size_t)out.C * out.H * out.W;
    k_fma3<<<gcfg(n), 256, 0, c.stream>>>(kernel_wrap.p, alpha.p, beta.p, out.p, n);
    return out;
}

} // namespace

// ================================ entry ====================================
extern "C" void kernel_launch(void* const* d_in, const int* in_sizes, int n_in,
                              void* d_out, int out_size, void* d_ws, size_t ws_size,
                              hipStream_t stream) {
    (void)in_sizes; (void)out_size;
    Ctx c{(char*)d_ws, ws_size, 0, stream};

    Tensor img_blur        {(float*)d_in[0], 3, 256, 256};
    Tensor last_img_blur   {(float*)d_in[1], 3, 256, 256};
    Tensor output_last_img {(float*)d_in[2], 3, 256, 256};
    Tensor out_last_fea    {(float*)d_in[3], 256, 64, 64};
    Tensor out_last_fea_dn {(float*)d_in[4], 256, 32, 32};

    Fetch F{d_in, n_in, 5};
    Params P;
    P.img_down = F.cv(3, 3, 3); P.down_fea_1 = F.cv(256, 256, 3); P.fea = F.cv(256, 128, 3);
    P.conv1_1 = F.cv(3, 32, 3);  P.conv1_2 = F.res(32);  P.conv1_3 = F.res(32);
    P.conv2_1 = F.cv(32, 64, 3); P.conv2_2 = F.res(64);  P.conv2_3 = F.res(64);
    P.conv3_1 = F.cv(64, 128, 3);P.conv3_2 = F.res(128); P.conv3_3 = F.res(128);
    P.kconv1_1 = F.cv(9, 32, 3);  P.kconv1_2 = F.res(32);  P.kconv1_3 = F.res(32);
    P.kconv2_1 = F.cv(32, 64, 3); P.kconv2_2 = F.res(64);  P.kconv2_3 = F.res(64);
    P.kconv3_1 = F.cv(64, 128, 3);P.kconv3_2 = F.res(128); P.kconv3_3 = F.res(128);
    P.fac_warp = F.fac(128); P.fac_deblur = F.fac(256);
    P.kconv4 = F.cv(3200, 128, 1);
    P.upconv2_u = F.cv(256, 64, 4); P.upconv2_2 = F.res(64); P.upconv2_1 = F.res(64);
    P.upconv1_u = F.cv(64, 32, 4);  P.upconv1_2 = F.res(32); P.upconv1_1 = F.res(32);
    P.img_prd = F.cv(32, 3, 3);
    P.cmap = F.cmap();

    // ---- downsampled inputs (plain conv2d, no lrelu) ----
    Tensor img_blur_dn = conv(c, img_blur, P.img_down, 2, 1, false);
    Tensor last_dn     = conv(c, last_img_blur, P.img_down, 2, 1, false);
    Tensor out_img_dn  = conv(c, output_last_img, P.img_down, 2, 1, false);

    Tensor m3[3] = {img_blur, last_img_blur, output_last_img};
    Tensor merge = concatN(c, m3, 3);                       // 9ch 256^2
    Tensor m3d[3] = {img_blur_dn, last_dn, out_img_dn};
    Tensor merge_dn = concatN(c, m3d, 3);                   // 9ch 128^2

    auto kenc = [&](Tensor x) {
        Tensor h = conv(c, x, P.kconv1_1);
        h = resb(c, h, P.kconv1_2); h = resb(c, h, P.kconv1_3);
        h = conv(c, h, P.kconv2_1, 2);
        h = resb(c, h, P.kconv2_2); h = resb(c, h, P.kconv2_3);
        h = conv(c, h, P.kconv3_1, 2);
        h = resb(c, h, P.kconv3_2); h = resb(c, h, P.kconv3_3);
        return h;
    };
    Tensor kconv3    = kenc(merge);       // 128ch 64^2
    Tensor kconv3_dn = kenc(merge_dn);    // 128ch 32^2

    Tensor kernel_warp     = fac_run(c, kconv3, P.fac_warp);     // 3200ch 64^2
    Tensor kernel_warp_ori = kernel_warp;
    Tensor kernel_warp_dn  = fac_run(c, kconv3_dn, P.fac_warp);  // 3200ch 32^2

    auto enc = [&](Tensor x) {
        Tensor h = conv(c, x, P.conv1_1);
        h = resb(c, h, P.conv1_2); h = resb(c, h, P.conv1_3);
        h = conv(c, h, P.conv2_1, 2);
        h = resb(c, h, P.conv2_2); h = resb(c, h, P.conv2_3);
        h = conv(c, h, P.conv3_1, 2);
        h = resb(c, h, P.conv3_2); h = resb(c, h, P.conv3_3);
        return h;
    };
    Tensor last_fea    = enc(last_img_blur);
    Tensor img_fea     = enc(img_blur);
    Tensor last_fea_dn = enc(last_dn);
    Tensor img_fea_dn  = enc(img_blur_dn);

    kernel_warp    = confidence(c, P.cmap, img_blur, last_img_blur, img_fea, last_fea,
                                out_last_fea, kernel_warp);
    kernel_warp_dn = confidence(c, P.cmap, img_blur_dn, last_dn, img_fea_dn, last_fea_dn,
                                out_last_fea_dn, kernel_warp_dn);

    // kernel_warp += upsample2(kernel_warp_dn)
    {
        Tensor fused = t_alloc(c, 3200, 64, 64);
        size_t n = (size_t)3200 * 64 * 64;
        k_add_up<<<gcfg(n), 256, 0, stream>>>(kernel_warp.p, kernel_warp_dn.p, fused.p, 3200, 64, 64);
        kernel_warp = fused;
    }

    Tensor kconv4 = conv(c, kernel_warp, P.kconv4);              // 128ch
    Tensor kernel_deblur = fac_run(c, concat2(c, kconv3, kconv4), P.fac_deblur);

    // ---- outputs: d_out = [output_img(3*256^2) | conv3(256*64^2) | out_fea_down(256*32^2)]
    float* out_img_p  = (float*)d_out;
    float* conv3_p    = out_img_p + (size_t)3 * 256 * 256;
    float* out_fead_p = conv3_p + (size_t)256 * 64 * 64;

    size_t nfeat = (size_t)128 * 64 * 64;
    k_kconv5<<<gcfg(nfeat), 256, 0, stream>>>(img_fea.p, kernel_deblur.p, conv3_p, 128, 64, 64, 0);
    Tensor olf = conv(c, out_last_fea, P.fea);                   // 128ch
    k_kconv5<<<gcfg(nfeat), 256, 0, stream>>>(olf.p, kernel_warp.p,     conv3_p + nfeat, 128, 64, 64, 0);
    k_kconv5<<<gcfg(nfeat), 256, 0, stream>>>(olf.p, kernel_warp_ori.p, conv3_p + nfeat, 128, 64, 64, 1);
    Tensor conv3{conv3_p, 256, 64, 64};

    // decoder
    Tensor up2 = t_alloc(c, 64, 128, 128);
    {
        size_t n = (size_t)64 * 128 * 128;
        k_tconv4x2<<<gcfg(n), 256, 0, stream>>>(conv3.p, P.upconv2_u.w, P.upconv2_u.b, up2.p, 256, 64, 64, 64);
    }
    up2 = resb(c, up2, P.upconv2_2); up2 = resb(c, up2, P.upconv2_1);
    Tensor up1 = t_alloc(c, 32, 256, 256);
    {
        size_t n = (size_t)32 * 256 * 256;
        k_tconv4x2<<<gcfg(n), 256, 0, stream>>>(up2.p, P.upconv1_u.w, P.upconv1_u.b, up1.p, 64, 128, 128, 32);
    }
    up1 = resb(c, up1, P.upconv1_2); up1 = resb(c, up1, P.upconv1_1);

    conv(c, up1, P.img_prd, 1, -1, true, img_blur.p, out_img_p);         // lrelu(conv)+img_blur
    conv(c, conv3, P.down_fea_1, 2, 1, false, nullptr, out_fead_p);      // out_fea_down
}
